// Gemma3Attention_55688545960175
// MI455X (gfx1250) — matrix operand order repaired and checked by reading, not compiled, not run on hardware
//
#include <hip/hip_runtime.h>

#define DEV __device__ __forceinline__

typedef _Float16 h8_t  __attribute__((ext_vector_type(8)));
typedef _Float16 h16_t __attribute__((ext_vector_type(16)));
typedef float    f8_t  __attribute__((ext_vector_type(8)));
typedef float    f4_t  __attribute__((ext_vector_type(4)));

constexpr int BB   = 2;
constexpr int TT   = 2048;
constexpr int HID  = 3840;
constexpr int NH   = 16;    // query heads
constexpr int NKV  = 8;     // kv heads
constexpr int DH   = 256;   // head dim
constexpr int WIN  = 1024;  // sliding window
constexpr int NQKV = NH*DH + 2*NKV*DH; // 8192

// ---------------------------------------------------------------- WMMA helper
DEV f8_t wmma32(h16_t a, h16_t b, f8_t c) {
    // D = A(16x32 f16) * B(32x16 f16) + C(16x16 f32)
    return __builtin_amdgcn_wmma_f32_16x16x32_f16(
        false, a, false, b, (short)0, c, false, false);
}

// A fragment (16x32, 16-bit): lane L -> row L&15;
// elements 0..7  : K = k0 + 16*(L>>4) + e       (contiguous 16B)
// elements 8..15 : K = k0 + 16*(L>>4) + 8 + e   (contiguous 16B)
DEV h16_t load_a_frag(const _Float16* __restrict__ row, int k0, int g) {
    h8_t lo = *(const h8_t*)(row + k0 + 16*g);
    h8_t hi = *(const h8_t*)(row + k0 + 16*g + 8);
    h16_t r;
#pragma unroll
    for (int i = 0; i < 8; ++i) { r[i] = lo[i]; r[i+8] = hi[i]; }
    return r;
}

DEV f8_t zero8() {
    f8_t z = {0.f,0.f,0.f,0.f,0.f,0.f,0.f,0.f};
    return z;
}

// Async global->LDS copy of 16B per lane (ASYNCcnt-tracked DMA path).
// VDST vgpr = LDS byte offset (low 32 bits of flat shared address),
// VADDR vgpr pair = 64-bit global address, GV mode (saddr = off).
DEV void async_copy16(const _Float16* g, _Float16* l) {
    unsigned loff = (unsigned)(unsigned long long)(uintptr_t)l;
    unsigned long long ga = (unsigned long long)(uintptr_t)g;
    asm volatile("global_load_async_to_lds_b128 %0, %1, off"
                 :: "v"(loff), "v"(ga) : "memory");
}
DEV void wait_async0() { asm volatile("s_wait_asynccnt 0" ::: "memory"); }

// ---------------------------------------------------------------- 1) fp32->f16
__global__ __launch_bounds__(256) void cvt_f32_f16_kernel(
    const float* __restrict__ src, _Float16* __restrict__ dst, int n)
{
    int i = (blockIdx.x * 256 + threadIdx.x) * 8;
    if (i >= n) return;
    f4_t a = *(const f4_t*)(src + i);
    f4_t b = *(const f4_t*)(src + i + 4);
    h8_t o;
#pragma unroll
    for (int j = 0; j < 4; ++j) { o[j] = (_Float16)a[j]; o[j+4] = (_Float16)b[j]; }
    *(h8_t*)(dst + i) = o;
}

// ---------------------------------------------------------------- 2/5) GEMM
// C[M,N](f32) = A[M,K](f16) @ Bw[N,K]^T(f16).
// 256 thr = 8 waves; block tile 64x128; wave tile 32x32 (4 WMMA accums).
// K-panels (A 64x32, B 128x32) double-buffered in LDS, filled with
// global_load_async_to_lds_b128 so the DMA of step k+1 overlaps WMMA of step k.
__global__ __launch_bounds__(256, 2) void gemm_f16_wmma_async_kernel(
    const _Float16* __restrict__ A, const _Float16* __restrict__ Bw,
    float* __restrict__ C, int M, int N, int K)
{
    __shared__ _Float16 shA[2][64  * 32];   // 2 x 4 KB
    __shared__ _Float16 shB[2][128 * 32];   // 2 x 8 KB

    const int tid  = threadIdx.x;
    const int lane = tid & 31;
    const int wave = tid >> 5;
    const int ln = lane & 15, g = lane >> 4;
    const int m_blk = blockIdx.y * 64;
    const int n_blk = blockIdx.x * 128;
    const int m_w = (wave >> 2) * 32;       // wave tile inside block
    const int n_w = (wave & 3) * 32;

    auto stage = [&](int buf, int k0) {
        // A panel: 64 rows x 32 halves = 256 x 16B chunks, 1 per thread
        {
            int row = tid >> 2, c = tid & 3;
            async_copy16(A + (size_t)(m_blk + row) * K + k0 + c * 8,
                         &shA[buf][row * 32 + c * 8]);
        }
        // B panel: 128 rows x 32 halves = 512 x 16B chunks, 2 per thread
#pragma unroll
        for (int r = 0; r < 2; ++r) {
            int idx = tid + r * 256;
            int row = idx >> 2, c = idx & 3;
            async_copy16(Bw + (size_t)(n_blk + row) * K + k0 + c * 8,
                         &shB[buf][row * 32 + c * 8]);
        }
    };

    f8_t c00 = zero8(), c01 = zero8(), c10 = zero8(), c11 = zero8();

    stage(0, 0);
    wait_async0();
    __syncthreads();

    for (int k0 = 0; k0 < K; k0 += 32) {
        const int buf = (k0 >> 5) & 1;
        if (k0 + 32 < K) stage(buf ^ 1, k0 + 32);  // async DMA next panel

        h16_t af0 = load_a_frag(&shA[buf][(m_w + ln)      * 32], 0, g);
        h16_t af1 = load_a_frag(&shA[buf][(m_w + 16 + ln) * 32], 0, g);
        h16_t bf0 = *(const h16_t*)(&shB[buf][(n_w + ln)      * 32] + 16 * g);
        h16_t bf1 = *(const h16_t*)(&shB[buf][(n_w + 16 + ln) * 32] + 16 * g);

        c00 = wmma32(af0, bf0, c00);
        c01 = wmma32(af0, bf1, c01);
        c10 = wmma32(af1, bf0, c10);
        c11 = wmma32(af1, bf1, c11);

        wait_async0();       // own async stores complete
        __syncthreads();     // panel visible to all waves / safe to reuse buf
    }

#pragma unroll
    for (int j = 0; j < 8; ++j) {
        int r0 = m_blk + m_w + j + 8*g, r1 = r0 + 16;
        int q0 = n_blk + n_w + ln,      q1 = q0 + 16;
        C[(size_t)r0 * N + q0] = c00[j];
        C[(size_t)r0 * N + q1] = c01[j];
        C[(size_t)r1 * N + q0] = c10[j];
        C[(size_t)r1 * N + q1] = c11[j];
    }
}

// ---------------------------------------------------------------- 3) norm+rope
// One wave per (b,t,unit). units: 0..15 = q heads, 16..23 = k heads, 24..31 = v.
// Writes q[B,NH,T,D] (normed, roped, *1/16), k[B,NKV,T,D] (normed, roped),
// vT[B,NKV,D,T].
__global__ __launch_bounds__(128) void norm_rope_kernel(
    const float* __restrict__ qkv, const float* __restrict__ cosl,
    const float* __restrict__ sinl, const float* __restrict__ qw,
    const float* __restrict__ kw,
    _Float16* __restrict__ qb, _Float16* __restrict__ kb,
    _Float16* __restrict__ vt)
{
    const int bt   = blockIdx.x;
    const int b    = bt >> 11;           // T = 2048
    const int t    = bt & (TT - 1);
    const int wave = threadIdx.x >> 5;
    const int lane = threadIdx.x & 31;
    const int d    = lane * 8;           // 8 contiguous dims per lane

    for (int u = wave; u < 32; u += 4) {
        const float* src = qkv + (size_t)bt * NQKV + u * DH;
        f4_t x0 = *(const f4_t*)(src + d);
        f4_t x1 = *(const f4_t*)(src + d + 4);
        float xv[8];
#pragma unroll
        for (int i = 0; i < 4; ++i) { xv[i] = x0[i]; xv[i+4] = x1[i]; }

        float ss = 0.f;
#pragma unroll
        for (int i = 0; i < 8; ++i) ss += xv[i] * xv[i];
#pragma unroll
        for (int off = 1; off < 32; off <<= 1) ss += __shfl_xor(ss, off, 32);
        float rn = rsqrtf(ss * (1.0f / DH) + 1e-6f);

        if (u < 24) {                      // q or k: norm + rope
            const float* nw = (u < 16) ? qw : kw;
            float nv[8];
#pragma unroll
            for (int i = 0; i < 8; ++i) nv[i] = xv[i] * rn * nw[d + i];
            float rv[8];
#pragma unroll
            for (int i = 0; i < 8; ++i) {
                float oth = __shfl_xor(nv[i], 16, 32);   // rope partner (d ^ 128)
                int fi = (d & 127) + i;
                float c = cosl[t * 128 + fi];
                float s = sinl[t * 128 + fi];
                rv[i] = (lane < 16) ? (nv[i] * c - oth * s)
                                    : (nv[i] * c + oth * s);
            }
            h8_t o;
            if (u < 16) {
#pragma unroll
                for (int i = 0; i < 8; ++i) o[i] = (_Float16)(rv[i] * 0.0625f); // fold 1/sqrt(D)
                *(h8_t*)(qb + (((size_t)b * NH + u) * TT + t) * DH + d) = o;
            } else {
#pragma unroll
                for (int i = 0; i < 8; ++i) o[i] = (_Float16)rv[i];
                *(h8_t*)(kb + (((size_t)b * NKV + (u - 16)) * TT + t) * DH + d) = o;
            }
        } else {                           // v: transpose store
            int hv = u - 24;
#pragma unroll
            for (int i = 0; i < 8; ++i)
                vt[(((size_t)b * NKV + hv) * DH + d + i) * TT + t] = (_Float16)xv[i];
        }
    }
}

// ---------------------------------------------------------------- 4) attention
// 4 waves/block, each wave owns 16 query rows. Computes S^T = K @ Q^T so the
// softmaxed C-tile is directly the A-fragment of P for the P@V WMMA.
__global__ __launch_bounds__(128, 1) void attn_wmma_kernel(
    const _Float16* __restrict__ qb, const _Float16* __restrict__ kb,
    const _Float16* __restrict__ vtb, _Float16* __restrict__ ob)
{
    __shared__ _Float16 ldsq[4][16 * DH];   // 32 KB: Q tile per wave

    const int wave = threadIdx.x >> 5;
    const int lane = threadIdx.x & 31;
    const int ln = lane & 15, g = lane >> 4;
    const int q0 = blockIdx.x * 64 + wave * 16;
    const int h  = blockIdx.y;
    const int b  = blockIdx.z;
    const int kvh = h >> 1;                 // GQA: 2 q-heads per kv-head

    const _Float16* qrow = qb  + ((size_t)(b * NH  + h)   * TT) * DH;
    const _Float16* krow = kb  + ((size_t)(b * NKV + kvh) * TT) * DH;
    const _Float16* vtp  = vtb + ((size_t)(b * NKV + kvh) * DH) * TT;

    // stage this wave's 16x256 Q tile into LDS (ds_store_b128)
#pragma unroll
    for (int c = 0; c < 8; ++c) {
        int idx = c * 32 + lane;            // 256 chunks of 32B
        int row = idx >> 4, col = idx & 15;
        *(h16_t*)&ldsq[wave][row * DH + col * 16] =
            *(const h16_t*)(qrow + (size_t)(q0 + row) * DH + col * 16);
    }

    f8_t acc[16];
#pragma unroll
    for (int dt = 0; dt < 16; ++dt) acc[dt] = zero8();
    float m_run = -1e30f, l_run = 0.f;

    const int q  = q0 + ln;
    int kstart = q0 - WIN + 1; if (kstart < 0) kstart = 0;
    kstart &= ~31;
    const int kend = q0 + 15;

    for (int k0 = kstart; k0 <= kend; k0 += 32) {
        // ---- S^T = K_tile(32x256) @ Q^T(256x16)
        f8_t s0 = zero8(), s1 = zero8();
#pragma unroll
        for (int ds = 0; ds < 8; ++ds) {
            h16_t kf0 = load_a_frag(krow + (size_t)(k0 + ln)      * DH, ds * 32, g);
            h16_t kf1 = load_a_frag(krow + (size_t)(k0 + 16 + ln) * DH, ds * 32, g);
            h16_t qf  = *(const h16_t*)&ldsq[wave][ln * DH + ds * 32 + 16 * g];
            s0 = wmma32(kf0, qf, s0);
            s1 = wmma32(kf1, qf, s1);
        }

        // ---- mask + online softmax (stats per q-col; one shfl_xor(16))
        float sv0[8], sv1[8];
        bool  vd0[8], vd1[8];
#pragma unroll
        for (int j = 0; j < 8; ++j) {
            int kk0 = k0 + j + 8 * g;
            int kk1 = kk0 + 16;
            vd0[j] = (kk0 <= q) && (q - kk0 < WIN);
            vd1[j] = (kk1 <= q) && (q - kk1 < WIN);
            sv0[j] = vd0[j] ? s0[j] : -1e30f;
            sv1[j] = vd1[j] ? s1[j] : -1e30f;
        }
        float mt = -1e30f;
#pragma unroll
        for (int j = 0; j < 8; ++j) mt = fmaxf(mt, fmaxf(sv0[j], sv1[j]));
        mt = fmaxf(mt, __shfl_xor(mt, 16, 32));
        float m_new = fmaxf(m_run, mt);
        float sc_old = __expf(m_run - m_new);

        float p0[8], p1[8], lt = 0.f;
#pragma unroll
        for (int j = 0; j < 8; ++j) {
            p0[j] = vd0[j] ? __expf(sv0[j] - m_new) : 0.f;
            p1[j] = vd1[j] ? __expf(sv1[j] - m_new) : 0.f;
            lt += p0[j] + p1[j];
        }
        lt += __shfl_xor(lt, 16, 32);
        l_run = l_run * sc_old + lt;
        m_run = m_new;

        // rescale O rows (row j+8g stat lives in lane 8g+j)
#pragma unroll
        for (int j = 0; j < 8; ++j) {
            float fac = __shfl(sc_old, 8 * g + j, 32);
#pragma unroll
            for (int dt = 0; dt < 16; ++dt) acc[dt][j] *= fac;
        }

        // P fragment: S^T C-layout == P A-layout, just convert to f16
        h16_t pf;
#pragma unroll
        for (int j = 0; j < 8; ++j) {
            pf[j]     = (_Float16)p0[j];
            pf[j + 8] = (_Float16)p1[j];
        }

        // ---- O += P(16x32) @ V(32x256); V^T rows give contiguous B-fragments
#pragma unroll
        for (int dt = 0; dt < 16; ++dt) {
            h8_t vlo = *(const h8_t*)(vtp + (size_t)(dt * 16 + ln) * TT + k0 + 8 * g);
            h8_t vhi = *(const h8_t*)(vtp + (size_t)(dt * 16 + ln) * TT + k0 + 16 + 8 * g);
            h16_t vf = __builtin_shufflevector(vlo, vhi, 0,1,2,3,4,5,6,7,8,9,10,11,12,13,14,15);
            acc[dt] = wmma32(pf, vf, acc[dt]);
        }
    }

    // ---- epilogue: divide by l, write [B*T, NH*DH] f16 for out-proj
#pragma unroll
    for (int j = 0; j < 8; ++j) {
        float inv = 1.0f / __shfl(l_run, 8 * g + j, 32);
        int r = q0 + j + 8 * g;
        size_t rbase = (size_t)(b * TT + r) * (NH * DH) + h * DH;
#pragma unroll
        for (int dt = 0; dt < 16; ++dt)
            ob[rbase + dt * 16 + ln] = (_Float16)(acc[dt][j] * inv);
    }
}

// ---------------------------------------------------------------- launch
extern "C" void kernel_launch(void* const* d_in, const int* in_sizes, int n_in,
                              void* d_out, int out_size, void* d_ws, size_t ws_size,
                              hipStream_t stream) {
    const float* x    = (const float*)d_in[0];
    const float* cosl = (const float*)d_in[3];
    const float* sinl = (const float*)d_in[4];
    const float* wq   = (const float*)d_in[6];
    const float* wk   = (const float*)d_in[7];
    const float* wv   = (const float*)d_in[8];
    const float* wo   = (const float*)d_in[9];
    const float* qnw  = (const float*)d_in[10];
    const float* knw  = (const float*)d_in[11];
    float* out = (float*)d_out;

    const int M = BB * TT;                          // 4096
    const size_t nX  = (size_t)M * HID;             // 15,728,640
    const size_t nWq = (size_t)NH  * DH * HID;      // 15,728,640
    const size_t nWk = (size_t)NKV * DH * HID;      //  7,864,320
    const size_t nWo = (size_t)HID * NH * DH;       // 15,728,640

    size_t off = 0;
    auto carve = [&](size_t bytes) {
        size_t o = off;
        off += (bytes + 255) & ~(size_t)255;
        return (char*)d_ws + o;
    };
    _Float16* x16   = (_Float16*)carve(nX * 2);
    _Float16* w16   = (_Float16*)carve((size_t)NQKV * HID * 2);    // wq|wk|wv
    _Float16* wo16  = (_Float16*)carve(nWo * 2);
    float*    qkv32 = (float*)   carve((size_t)M * NQKV * 4);
    _Float16* qb    = (_Float16*)carve((size_t)BB * NH  * TT * DH * 2);
    _Float16* kb    = (_Float16*)carve((size_t)BB * NKV * TT * DH * 2);
    _Float16* vt    = (_Float16*)carve((size_t)BB * NKV * DH * TT * 2);
    _Float16* ob    = (_Float16*)carve((size_t)M * NH * DH * 2);
    (void)ws_size; (void)n_in; (void)in_sizes; (void)out_size;

    auto cvt = [&](const float* s, _Float16* d, size_t n) {
        int blocks = (int)((n / 8 + 255) / 256);
        cvt_f32_f16_kernel<<<blocks, 256, 0, stream>>>(s, d, (int)n);
    };
    cvt(x,  x16, nX);
    cvt(wq, w16,                 nWq);
    cvt(wk, w16 + nWq,           nWk);
    cvt(wv, w16 + nWq + nWk,     nWk);
    cvt(wo, wo16, nWo);

    // QKV projection: [4096 x 3840] @ [8192 x 3840]^T -> f32
    gemm_f16_wmma_async_kernel<<<dim3(NQKV / 128, M / 64), 256, 0, stream>>>(
        x16, w16, qkv32, M, NQKV, HID);

    // RMSNorm + RoPE + layout change
    norm_rope_kernel<<<dim3(M), 128, 0, stream>>>(
        qkv32, cosl, sinl, qnw, knw, qb, kb, vt);

    // Sliding-window flash attention
    attn_wmma_kernel<<<dim3(TT / 64, NH, BB), 128, 0, stream>>>(qb, kb, vt, ob);

    // Output projection: [4096 x 4096] @ [3840 x 4096]^T -> f32 d_out
    gemm_f16_wmma_async_kernel<<<dim3(HID / 128, M / 64), 256, 0, stream>>>(
        ob, wo16, out, M, HID, NH * DH);
}
